// RawProposalGenerator_43757126812083
// MI455X (gfx1250) — matrix operand order repaired and checked by reading, not compiled, not run on hardware
//
#include <hip/hip_runtime.h>
#include <hip/hip_bf16.h>
#include <stdint.h>

typedef __attribute__((ext_vector_type(16))) __bf16        v16bf;
typedef __attribute__((ext_vector_type(8)))  float         v8f;
typedef __attribute__((ext_vector_type(8)))  unsigned int  v8u;   // 32B raw

namespace {
constexpr int BS  = 16;
constexpr int NQ  = 100;
constexpr int C   = 256;
constexpr int H   = 128;
constexpr int W   = 128;
constexpr int HW  = H * W;
constexpr int NTN = (NQ + 15) / 16;   // 7 n-tiles (rows 100..111 zero-padded)
constexpr int NCH = C / 32;           // 8 c-chunks
constexpr int WAVES = 8;              // waves per block in GEMM kernel
// packed-A record: 32 lanes x 16 bf16 = 512 bf16 elements (1KB)
constexpr int RECE = 512;             // elements per plane record
}

// ---------------------------------------------------------------------------
// Kernel 0: pre-split S (support) into bf16 hi/lo planes, laid out exactly as
// the CDNA5 16-bit A-matrix fragment (per-lane 16 contiguous bf16).
// Layout: pA[b][ch][t][plane(hi=0,lo=1)][lane][16]
// ---------------------------------------------------------------------------
__global__ __launch_bounds__(256) void pack_a(const float* __restrict__ S,   // [NQ,BS,C]
                                              __bf16* __restrict__ pA)
{
  const int lane = threadIdx.x & 31;
  const int wave = threadIdx.x >> 5;
  const int b    = blockIdx.y;
  const int rec  = blockIdx.x * 8 + wave;   // 0..55
  const int ch   = rec & 7;
  const int t    = rec >> 3;
  const int half = lane >> 4;
  const int l15  = lane & 15;
  const int c0   = ch * 32;
  const int kbA  = half * 8;                // A layout: lanes16-31 start at K=8

  const int   n  = t * 16 + l15;
  const int   nn = (n < NQ) ? n : (NQ - 1);
  const float mk = (n < NQ) ? 1.f : 0.f;

  const float* srow = S + ((size_t)nn * BS + b) * C + c0 + kbA;
  const float4* p0 = (const float4*)srow;          // K = kbA + 0..7
  const float4* p1 = (const float4*)(srow + 16);   // K = kbA + 16..23
  float4 a0 = p0[0], a1 = p0[1], a2 = p1[0], a3 = p1[1];
  float ta[16];
  ta[0]=a0.x;  ta[1]=a0.y;  ta[2]=a0.z;  ta[3]=a0.w;
  ta[4]=a1.x;  ta[5]=a1.y;  ta[6]=a1.z;  ta[7]=a1.w;
  ta[8]=a2.x;  ta[9]=a2.y;  ta[10]=a2.z; ta[11]=a2.w;
  ta[12]=a3.x; ta[13]=a3.y; ta[14]=a3.z; ta[15]=a3.w;

  v16bf ah, al;
#pragma unroll
  for (int e = 0; e < 16; ++e) {
    float v = ta[e] * mk;
    __bf16 hh = (__bf16)v;
    ah[e] = hh;
    al[e] = (__bf16)(v - (float)hh);
  }

  const size_t base = ((((size_t)b * NCH + ch) * NTN + t) * 2) * RECE + (size_t)lane * 16;
  *(v8u*)(pA + base)        = __builtin_bit_cast(v8u, ah);
  *(v8u*)(pA + base + RECE) = __builtin_bit_cast(v8u, al);
}

// ---------------------------------------------------------------------------
// Kernel 1: sim[b,n,k] = sum_c S[n,b,c] * Q[k,b,c]
// Split-precision bf16 GEMM: x = hi(x)+lo(x); x*y ~= hi*hi + hi*lo + lo*hi.
// One wave owns 16 k-columns and all 7 n-tiles -> Q read once from HBM;
// A fragments come pre-split/pre-laid-out from d_ws (L2-resident).
// ---------------------------------------------------------------------------
__global__ __launch_bounds__(256) void sim_gemm(const float*  __restrict__ Q,   // [HW,BS,C]
                                                const __bf16* __restrict__ pA,  // packed A
                                                float* __restrict__ sim)        // [BS,NQ,HW]
{
  const int lane = threadIdx.x & 31;
  const int wave = threadIdx.x >> 5;
  const int b    = blockIdx.y;
  const int k0   = (blockIdx.x * WAVES + wave) * 16;
  const int half = lane >> 4;
  const int l15  = lane & 15;

  // B operand source (Q^T tile): column N = l15, K = half*8 + e (e<8), 16 + half*8 + (e-8) (e>=8)
  const float* qrow = Q + ((size_t)(k0 + l15) * BS + b) * C + half * 8;
  const __bf16* pAb = pA + ((size_t)b * NCH * NTN * 2) * RECE + (size_t)lane * 16;

  v8f acc[NTN] = {};                 // fp32 accumulators, C/D layout

  for (int ch = 0; ch < NCH; ++ch) {
    const int c0 = ch * 32;
    // ---- B fragment: 2x8 floats per lane (K = half*8 + 0..7, half*8 + 16..23), split hi/lo ----
    float tb[16];
    {
      const float4* p = (const float4*)(qrow + c0);
      float4 f0 = p[0], f1 = p[1], f2 = p[4], f3 = p[5];
      tb[0]=f0.x; tb[1]=f0.y; tb[2]=f0.z; tb[3]=f0.w;
      tb[4]=f1.x; tb[5]=f1.y; tb[6]=f1.z; tb[7]=f1.w;
      tb[8]=f2.x; tb[9]=f2.y; tb[10]=f2.z; tb[11]=f2.w;
      tb[12]=f3.x; tb[13]=f3.y; tb[14]=f3.z; tb[15]=f3.w;
    }
    v16bf bh, bl;
#pragma unroll
    for (int e = 0; e < 16; ++e) {
      __bf16 hh = (__bf16)tb[e];
      bh[e] = hh;
      bl[e] = (__bf16)(tb[e] - (float)hh);
    }

    const __bf16* pAc = pAb + ((size_t)ch * NTN * 2) * RECE;
#pragma unroll
    for (int t = 0; t < NTN; ++t) {
      v16bf ah = __builtin_bit_cast(v16bf, *(const v8u*)(pAc + (size_t)t * 2 * RECE));
      v16bf al = __builtin_bit_cast(v16bf, *(const v8u*)(pAc + (size_t)t * 2 * RECE + RECE));
      acc[t] = __builtin_amdgcn_wmma_f32_16x16x32_bf16(false, ah, false, bh,
                                                       (short)0, acc[t], false, false);
      acc[t] = __builtin_amdgcn_wmma_f32_16x16x32_bf16(false, ah, false, bl,
                                                       (short)0, acc[t], false, false);
      acc[t] = __builtin_amdgcn_wmma_f32_16x16x32_bf16(false, al, false, bh,
                                                       (short)0, acc[t], false, false);
    }
  }

  // C/D layout: lane -> column k0+l15; VGPR i -> row i (lanes 0-15) / 8+i (16-31)
#pragma unroll
  for (int t = 0; t < NTN; ++t) {
#pragma unroll
    for (int i = 0; i < 8; ++i) {
      const int n = t * 16 + half * 8 + i;
      if (n < NQ)
        sim[((size_t)b * NQ + n) * HW + k0 + l15] = acc[t][i];
    }
  }
}

// ---------------------------------------------------------------------------
// Kernel 2: per (b,n): online softmax + argmax + coordinate expectations,
// then 3x3 peak-masked proposal. Reads sim exactly once.
// ---------------------------------------------------------------------------
__global__ __launch_bounds__(256) void softmax_coord(const float* __restrict__ sim,
                                                     float* __restrict__ loss,
                                                     float* __restrict__ prop)
{
  const int n = blockIdx.x;
  const int b = blockIdx.y;
  const float* row = sim + ((size_t)b * NQ + n) * HW;
  const int tid = threadIdx.x;

  float m = -INFINITY, s = 0.f, sx = 0.f, sy = 0.f;
  int bi = 0;
  for (int k = tid; k < HW; k += 256) {
    float v = row[k];
    if (v > m) {                      // strict > keeps first occurrence
      float r = expf(m - v);          // expf(-inf)=0 on first hit
      s *= r; sx *= r; sy *= r;
      m = v; bi = k;
    }
    float e = expf(v - m);
    float x = (float)(k & (W - 1)) + 0.5f;   // k % w
    float y = (float)(k >> 7)      + 0.5f;   // k / w
    s += e; sx = fmaf(e, x, sx); sy = fmaf(e, y, sy);
  }

  __shared__ float shm[256], shs[256], shx[256], shy[256];
  __shared__ int   shi[256];
  shm[tid]=m; shs[tid]=s; shx[tid]=sx; shy[tid]=sy; shi[tid]=bi;
  __syncthreads();
  for (int off = 128; off > 0; off >>= 1) {
    if (tid < off) {
      float m2 = shm[tid+off], s2 = shs[tid+off];
      float x2 = shx[tid+off], y2 = shy[tid+off];
      int   i2 = shi[tid+off];
      float M  = fmaxf(m, m2);
      float r1 = expf(m - M), r2 = expf(m2 - M);
      s  = s  * r1 + s2 * r2;
      sx = sx * r1 + x2 * r2;
      sy = sy * r1 + y2 * r2;
      if (m2 > m || (m2 == m && i2 < bi)) bi = i2;  // first-occurrence argmax
      m = M;
      shm[tid]=m; shs[tid]=s; shx[tid]=sx; shy[tid]=sy; shi[tid]=bi;
    }
    __syncthreads();
  }

  if (tid == 0) {
    const float inv_s = 1.f / s;
    const size_t o = ((size_t)b * NQ + n) * 2;
    loss[o + 0] = (sx * inv_s) * (1.f / W);   // / side_normalizer [w,h]
    loss[o + 1] = (sy * inv_s) * (1.f / H);

    // 3x3 window around argmax in the flat (128,128) interpretation
    const int r0 = bi >> 7, c0 = bi & (W - 1);
    float ls = 0.f, lsx = 0.f, lsy = 0.f;
    for (int dr = -1; dr <= 1; ++dr) {
      const int r = r0 + dr;
      if (r < 0 || r >= 128) continue;
      for (int dc = -1; dc <= 1; ++dc) {
        const int cc = c0 + dc;
        if (cc < 0 || cc >= 128) continue;
        const int k = (r << 7) | cc;
        const float e = expf(row[k] - m);
        ls  += e;
        lsx = fmaf(e, (float)cc + 0.5f, lsx);
        lsy = fmaf(e, (float)r  + 0.5f, lsy);
      }
    }
    const float denom = ls * inv_s + 1e-10f;
    prop[o + 0] = (lsx * inv_s) / denom * (1.f / W);
    prop[o + 1] = (lsy * inv_s) / denom * (1.f / H);
  }
}

extern "C" void kernel_launch(void* const* d_in, const int* in_sizes, int n_in,
                              void* d_out, int out_size, void* d_ws, size_t ws_size,
                              hipStream_t stream) {
  (void)in_sizes; (void)n_in; (void)out_size; (void)ws_size;
  const float* Q = (const float*)d_in[0];   // [16384,16,256] f32
  const float* S = (const float*)d_in[1];   // [100,16,256]   f32
  // d_in[2]=h, d_in[3]=w (device ints; shapes are static: 128x128)

  float* out  = (float*)d_out;
  float* loss = out;                                               // [16,100,2]
  float* sim  = out + (size_t)BS * NQ * 2;                         // [16,100,16384]
  float* prop = out + (size_t)BS * NQ * 2 + (size_t)BS * NQ * HW;  // [16,100,2]

  __bf16* pA = (__bf16*)d_ws;   // 16*8*7*2*512 bf16 = 1.75 MB packed A

  dim3 g0(NTN, BS);                         // 7x16 blocks of 8 waves: 56 recs/b
  pack_a<<<g0, 256, 0, stream>>>(S, pA);

  dim3 g1(HW / (16 * WAVES), BS);           // (128, 16) blocks of 8 waves
  sim_gemm<<<g1, 32 * WAVES, 0, stream>>>(Q, pA, sim);

  dim3 g2(NQ, BS);                          // (100, 16) blocks
  softmax_coord<<<g2, 256, 0, stream>>>(sim, loss, prop);
}
